// AFNO2D_10110353015398
// MI455X (gfx1250) — matrix operand order repaired and checked by reading, not compiled, not run on hardware
//
#include <hip/hip_runtime.h>
#include <hip/hip_bf16.h>

// ---------------------------------------------------------------------------
// AFNO2D on MI455X (gfx1250): DHT GEMMs via split-bf16 (hi/lo) WMMA with
// double-buffered LDS + vectorized b128 staging; fused block-MLP kernel.
// ---------------------------------------------------------------------------

typedef __attribute__((ext_vector_type(16))) __bf16 v16bf;
typedef __attribute__((ext_vector_type(8)))  float  v8f;

#define HH   180
#define WW   360
#define CC   512
#define NB   8
#define BS   64
#define MTOT (HH * WW)      // 64800 spatial positions (B=1)

struct __align__(16) F4    { float  v[4]; };
struct __align__(16) B16x8 { __bf16 v[8]; };
struct __align__(8)  B16x4 { __bf16 v[4]; };

// ---------------------------- WMMA helpers ---------------------------------

__device__ __forceinline__ v8f wmma_bf16(v16bf a, v16bf b, v8f c) {
  return __builtin_amdgcn_wmma_f32_16x16x32_bf16(
      false, a, false, b, (short)0, c, false, false);
}

// A fragment: row-major [M][K] in LDS.
// lane L: M = row0 + (L&15); chunk0 K = (L>>4)*8..+7 ; chunk1 K = 16+(L>>4)*8..+7
__device__ __forceinline__ v16bf load_a_frag(const __bf16* base, int ld,
                                             int row0, int kofs, int lane) {
  const __bf16* r = base + (row0 + (lane & 15)) * ld + kofs;
  int h8 = (lane >> 4) * 8;
  union { v16bf v; B16x8 b[2]; } u;
  u.b[0] = *(const B16x8*)(r + h8);
  u.b[1] = *(const B16x8*)(r + 16 + h8);
  return u.v;
}

// B fragment from B^T stored row-major [N][K] in LDS.
// lane L: N = n0 + (L&15); chunk0 K = (L>>4)*8..+7 ; chunk1 K = 16+(L>>4)*8..+7
__device__ __forceinline__ v16bf load_b_frag(const __bf16* base, int ld,
                                             int n0, int kofs, int lane) {
  const __bf16* r = base + (n0 + (lane & 15)) * ld + kofs + (lane >> 4) * 8;
  union { v16bf v; B16x8 b[2]; } u;
  u.b[0] = *(const B16x8*)(r);
  u.b[1] = *(const B16x8*)(r + 16);
  return u.v;
}

__device__ __forceinline__ void split_bf16(float v, __bf16& hi, __bf16& lo) {
  hi = (__bf16)v;
  lo = (__bf16)(v - (float)hi);
}

// ---------------------------- cas table ------------------------------------

__global__ void build_cas_kernel(float* __restrict__ out, int n) {
  long long idx = (long long)blockIdx.x * 256 + threadIdx.x;
  long long total = (long long)n * n;
  if (idx >= total) return;
  int i = (int)(idx / n), j = (int)(idx % n);
  int t = (int)(((long long)i * j) % n);          // exact angle reduction
  float a = 6.28318530717958647692f * ((float)t / (float)n);
  float s, c;
  __sincosf(a, &s, &c);
  out[idx] = c + s;
}

// ------------------- generic batched split-bf16 GEMM ------------------------

#define BM 128
#define BN 64
#define BK 32
#define LDS_LD 40   // 32 + 8 pad; multiple of 8 keeps 16B alignment

// A tile 128x32: thread t -> row t>>1, 16 contiguous K at (t&1)*16
__device__ __forceinline__ void stage_a_tile(const float* __restrict__ Ab,
                                             int lda, int M, int K, int m0, int k0,
                                             __bf16* aHi, __bf16* aLo, int t) {
  const int row = t >> 1;
  const int cb  = (t & 1) * 16;
  const int gm  = m0 + row;
  B16x8 h[2], l[2];
  if (k0 + BK <= K && gm < M) {
    const F4* p = (const F4*)(Ab + (long long)gm * lda + k0 + cb);
#pragma unroll
    for (int q = 0; q < 4; ++q) {
      F4 f = p[q];
#pragma unroll
      for (int j = 0; j < 4; ++j) {
        __bf16 hi, lo; split_bf16(f.v[j], hi, lo);
        h[q >> 1].v[(q & 1) * 4 + j] = hi;
        l[q >> 1].v[(q & 1) * 4 + j] = lo;
      }
    }
  } else {
#pragma unroll
    for (int j = 0; j < 16; ++j) {
      const int gk = k0 + cb + j;
      float v = (gm < M && gk < K) ? Ab[(long long)gm * lda + gk] : 0.0f;
      __bf16 hi, lo; split_bf16(v, hi, lo);
      h[j >> 3].v[j & 7] = hi;
      l[j >> 3].v[j & 7] = lo;
    }
  }
  B16x8* dh = (B16x8*)(aHi + row * LDS_LD + cb);
  B16x8* dl = (B16x8*)(aLo + row * LDS_LD + cb);
  dh[0] = h[0]; dh[1] = h[1];
  dl[0] = l[0]; dl[1] = l[1];
}

// Symmetric B (cas): B^T == B, read rows coalesced. thread t -> n = t>>2, 8 K at (t&3)*8
__device__ __forceinline__ void stage_b_sym(const float* __restrict__ Bb,
                                            int ldb, int N, int K, int n0, int k0,
                                            __bf16* bHi, __bf16* bLo, int t) {
  const int n  = t >> 2;
  const int kb = (t & 3) * 8;
  const int gn = n0 + n;
  B16x8 h, l;
  if (k0 + BK <= K && gn < N) {
    const F4* p = (const F4*)(Bb + (long long)gn * ldb + k0 + kb);
#pragma unroll
    for (int q = 0; q < 2; ++q) {
      F4 f = p[q];
#pragma unroll
      for (int j = 0; j < 4; ++j) split_bf16(f.v[j], h.v[q * 4 + j], l.v[q * 4 + j]);
    }
  } else {
#pragma unroll
    for (int j = 0; j < 8; ++j) {
      const int gk = k0 + kb + j;
      float v = (gn < N && gk < K) ? Bb[(long long)gn * ldb + gk] : 0.0f;
      split_bf16(v, h.v[j], l.v[j]);
    }
  }
  *(B16x8*)(bHi + n * LDS_LD + kb) = h;
  *(B16x8*)(bLo + n * LDS_LD + kb) = l;
}

// General B: transpose-gather; 64 consecutive threads read 64 consecutive columns.
__device__ __forceinline__ void stage_b_gen(const float* __restrict__ Bb,
                                            int ldb, int N, int K, int n0, int k0,
                                            __bf16* bHi, __bf16* bLo, int t) {
  const int n  = t & 63;
  const int kb = (t >> 6) * 8;
  const int gn = n0 + n;
  B16x8 h, l;
#pragma unroll
  for (int j = 0; j < 8; ++j) {
    const int gk = k0 + kb + j;
    float v = (gn < N && gk < K) ? Bb[(long long)gk * ldb + gn] : 0.0f;
    split_bf16(v, h.v[j], l.v[j]);
  }
  *(B16x8*)(bHi + n * LDS_LD + kb) = h;
  *(B16x8*)(bLo + n * LDS_LD + kb) = l;
}

__global__ __launch_bounds__(256) void gemm_split_bf16(
    const float* __restrict__ A, long long sA, int lda,
    const float* __restrict__ Bm, long long sB, int ldb,
    float* __restrict__ Cm, long long sC, int ldc,
    const float* __restrict__ R, long long sR,
    int M, int N, int K, float alpha, int bSym) {
  __shared__ __align__(16) __bf16 sAhi[2][BM * LDS_LD];
  __shared__ __align__(16) __bf16 sAlo[2][BM * LDS_LD];
  __shared__ __align__(16) __bf16 sBhi[2][BN * LDS_LD];
  __shared__ __align__(16) __bf16 sBlo[2][BN * LDS_LD];

  const int bz = blockIdx.z;
  const float* Ab = A + (long long)bz * sA;
  const float* Bb = Bm + (long long)bz * sB;
  float*       Cb = Cm + (long long)bz * sC;
  const float* Rb = R ? (R + (long long)bz * sR) : nullptr;

  const int m0 = blockIdx.x * BM;
  const int n0 = blockIdx.y * BN;
  const int t = threadIdx.x;
  const int lane = t & 31;
  const int wave = t >> 5;
  const int wm = wave & 3;   // 4 row-quadrants of 32
  const int wn = wave >> 2;  // 2 col-halves of 32

  v8f acc[2][2] = {};
  const int ksteps = (K + BK - 1) / BK;

  auto stage = [&](int ks, int buf) {
    const int k0 = ks * BK;
    stage_a_tile(Ab, lda, M, K, m0, k0, sAhi[buf], sAlo[buf], t);
    if (bSym) stage_b_sym(Bb, ldb, N, K, n0, k0, sBhi[buf], sBlo[buf], t);
    else      stage_b_gen(Bb, ldb, N, K, n0, k0, sBhi[buf], sBlo[buf], t);
  };

  stage(0, 0);
  __syncthreads();

  for (int ks = 0; ks < ksteps; ++ks) {
    const int cur = ks & 1;
    if (ks + 1 < ksteps) {
      stage(ks + 1, cur ^ 1);
      if (ks + 2 < ksteps) {   // CDNA5 global_prefetch_b8 two tiles ahead
        const int gm = m0 + (t >> 1);
        const int gk = (ks + 2) * BK + (t & 1) * 16;
        if (gm < M && gk < K) __builtin_prefetch(&Ab[(long long)gm * lda + gk], 0, 1);
      }
    }
#pragma unroll
    for (int mi = 0; mi < 2; ++mi) {
      const int r0 = wm * 32 + mi * 16;
      v16bf ahi = load_a_frag(sAhi[cur], LDS_LD, r0, 0, lane);
      v16bf alo = load_a_frag(sAlo[cur], LDS_LD, r0, 0, lane);
#pragma unroll
      for (int ni = 0; ni < 2; ++ni) {
        const int c0 = wn * 32 + ni * 16;
        v16bf bhi = load_b_frag(sBhi[cur], LDS_LD, c0, 0, lane);
        v16bf blo = load_b_frag(sBlo[cur], LDS_LD, c0, 0, lane);
        acc[mi][ni] = wmma_bf16(ahi, bhi, acc[mi][ni]);  // hi*hi
        acc[mi][ni] = wmma_bf16(alo, bhi, acc[mi][ni]);  // lo*hi
        acc[mi][ni] = wmma_bf16(ahi, blo, acc[mi][ni]);  // hi*lo
      }
    }
    __syncthreads();
  }

  // C layout: lane holds N = (lane&15); VGPR e -> M = e + (lane>>4)*8
#pragma unroll
  for (int mi = 0; mi < 2; ++mi) {
#pragma unroll
    for (int ni = 0; ni < 2; ++ni) {
      const int col = n0 + wn * 32 + ni * 16 + (lane & 15);
      const int rb  = m0 + wm * 32 + mi * 16 + (lane >> 4) * 8;
#pragma unroll
      for (int e = 0; e < 8; ++e) {
        const int row = rb + e;
        if (row < M && col < N) {
          float v = acc[mi][ni][e] * alpha;
          if (Rb) v += Rb[(long long)row * ldc + col];
          Cb[(long long)row * ldc + col] = v;
        }
      }
    }
  }
}

// ---------------- fused block-MLP (layer1 + layer2 + softshrink) ------------

#define MT  32
#define WLD 72   // 64 + 8 pad

__global__ __launch_bounds__(256) void afno_mlp_kernel(
    const float* __restrict__ xh,   // [MTOT][512] (DHT of x)
    const float* __restrict__ w1,   // [2][8][64][64]
    const float* __restrict__ b1,   // [2][8][64]
    const float* __restrict__ w2,   // [2][8][64][64]
    const float* __restrict__ b2,   // [2][8][64]
    float* __restrict__ ymid)       // [MTOT][512] softshrink output
{
  __shared__ __align__(16) __bf16 sW1a[BS * WLD], sW1b[BS * WLD];  // [o][i]
  __shared__ __align__(16) __bf16 sW2a[BS * WLD], sW2b[BS * WLD];
  __shared__ float  sB1k[BS], sB1n[BS], sB2k[BS], sB2n[BS];
  __shared__ __align__(16) __bf16 sXh[MT * WLD], sXn[MT * WLD];    // [m][i]
  __shared__ __align__(16) __bf16 sO1k[MT * WLD], sO1n[MT * WLD];  // [m][o]
  __shared__ __align__(16) __bf16 sO2k[MT * WLD];

  const int k  = blockIdx.y;
  const int p0 = blockIdx.x * MT;
  const int t  = threadIdx.x;
  const int lane = t & 31;
  const int wave = t >> 5;

  // ---- stage weights: w*a = w[0]+w[1] (transposed), w*b = w[0]-w[1]
  for (int idx = t; idx < BS * BS; idx += 256) {
    const int o = idx & 63, i = idx >> 6;
    const long long base1 = ((long long)k * BS + i) * BS + o;
    float a0 = w1[base1];
    float a1 = w1[(long long)NB * BS * BS + base1];
    sW1a[o * WLD + i] = (__bf16)(a0 + a1);
    sW1b[o * WLD + i] = (__bf16)(a0 - a1);
    float c0 = w2[base1];
    float c1 = w2[(long long)NB * BS * BS + base1];
    sW2a[o * WLD + i] = (__bf16)(c0 + c1);
    sW2b[o * WLD + i] = (__bf16)(c0 - c1);
  }
  if (t < BS) {
    sB1k[t] = b1[k * BS + t];
    sB1n[t] = b1[NB * BS + k * BS + t];
    sB2k[t] = b2[k * BS + t];
    sB2n[t] = b2[NB * BS + k * BS + t];
  }

  // ---- stage activations: xh tile + gathered xn tile (float4 loads)
#pragma unroll
  for (int it = 0; it < 2; ++it) {
    const int t2 = t + it * 256;      // 512 float4 slots = 32 rows x 16 f4
    const int mp = t2 >> 4;
    const int q  = t2 & 15;
    const int m  = p0 + mp;
    B16x4 bh, bn;
    if (m < MTOT) {
      const int h = m / WW, w = m % WW;
      const long long mn = (long long)((HH - h) % HH) * WW + ((WW - w) % WW);
      F4 a = *(const F4*)(xh + (long long)m * CC + k * BS + q * 4);
      F4 b = *(const F4*)(xh + mn * CC + k * BS + q * 4);
#pragma unroll
      for (int j = 0; j < 4; ++j) { bh.v[j] = (__bf16)a.v[j]; bn.v[j] = (__bf16)b.v[j]; }
    } else {
#pragma unroll
      for (int j = 0; j < 4; ++j) { bh.v[j] = (__bf16)0.0f; bn.v[j] = (__bf16)0.0f; }
    }
    *(B16x4*)(sXh + mp * WLD + q * 4) = bh;
    *(B16x4*)(sXn + mp * WLD + q * 4) = bn;
  }
  __syncthreads();

  // ---- layer 1: waves 0-3 -> o1k, waves 4-7 -> o1n; each wave: 1 m16 x 2 n16
  {
    const bool isN = wave >= 4;
    const int g = wave & 3;
    const int m16 = (g & 1) * 16;
    const int np  = (g >> 1) * 2;
    const __bf16* A1 = isN ? sXn : sXh;   // pairs with w1a
    const __bf16* A2 = isN ? sXh : sXn;   // pairs with w1b
    const float*  bv = isN ? sB1n : sB1k;
    __bf16* outL = isN ? sO1n : sO1k;
#pragma unroll
    for (int nn = 0; nn < 2; ++nn) {
      const int n16 = (np + nn) * 16;
      v8f acc = {};
#pragma unroll
      for (int kh = 0; kh < 2; ++kh) {
        acc = wmma_bf16(load_a_frag(A1, WLD, m16, kh * 32, lane),
                        load_b_frag(sW1a, WLD, n16, kh * 32, lane), acc);
        acc = wmma_bf16(load_a_frag(A2, WLD, m16, kh * 32, lane),
                        load_b_frag(sW1b, WLD, n16, kh * 32, lane), acc);
      }
      const int col = n16 + (lane & 15);
      const int rb  = m16 + (lane >> 4) * 8;
#pragma unroll
      for (int e = 0; e < 8; ++e) {
        float v = 0.5f * acc[e] + bv[col];
        v = v > 0.0f ? v : 0.0f;
        outL[(rb + e) * WLD + col] = (__bf16)v;
      }
    }
  }
  __syncthreads();

  // ---- layer 2, o2k: one 16x16 tile per wave
  const int m16 = (wave & 1) * 16;
  const int n16 = (wave >> 1) * 16;
  float o2kv[8];
  {
    v8f acc = {};
#pragma unroll
    for (int kh = 0; kh < 2; ++kh) {
      acc = wmma_bf16(load_a_frag(sO1k, WLD, m16, kh * 32, lane),
                      load_b_frag(sW2a, WLD, n16, kh * 32, lane), acc);
      acc = wmma_bf16(load_a_frag(sO1n, WLD, m16, kh * 32, lane),
                      load_b_frag(sW2b, WLD, n16, kh * 32, lane), acc);
    }
    const int col = n16 + (lane & 15);
    const int rb  = m16 + (lane >> 4) * 8;
#pragma unroll
    for (int e = 0; e < 8; ++e) {
      float v = 0.5f * acc[e] + sB2k[col];
      o2kv[e] = v;
      sO2k[(rb + e) * WLD + col] = (__bf16)v;
    }
  }
  __syncthreads();

  // ---- layer 2, o2n (consumes o2k per source bug); y = o2k + o2n; softshrink
  {
    v8f acc = {};
#pragma unroll
    for (int kh = 0; kh < 2; ++kh) {
      acc = wmma_bf16(load_a_frag(sO1n, WLD, m16, kh * 32, lane),
                      load_b_frag(sW2a, WLD, n16, kh * 32, lane), acc);
      acc = wmma_bf16(load_a_frag(sO2k, WLD, m16, kh * 32, lane),
                      load_b_frag(sW2b, WLD, n16, kh * 32, lane), acc);
    }
    const int col = n16 + (lane & 15);
    const int rb  = m16 + (lane >> 4) * 8;
#pragma unroll
    for (int e = 0; e < 8; ++e) {
      float y = o2kv[e] + (0.5f * acc[e] + sB2n[col]);
      float ay = fabsf(y) - 0.01f;                    // softshrink lambda
      float sv = ay > 0.0f ? ay : 0.0f;
      sv = y < 0.0f ? -sv : sv;
      const int m = p0 + rb + e;
      if (m < MTOT) ymid[(long long)m * CC + k * BS + col] = sv;
    }
  }
}

// ------------------------------- launch -------------------------------------

extern "C" void kernel_launch(void* const* d_in, const int* in_sizes, int n_in,
                              void* d_out, int out_size, void* d_ws, size_t ws_size,
                              hipStream_t stream) {
  (void)in_sizes; (void)n_in; (void)out_size; (void)ws_size;
  const float* x  = (const float*)d_in[0];
  const float* w1 = (const float*)d_in[1];
  const float* b1 = (const float*)d_in[2];
  const float* w2 = (const float*)d_in[3];
  const float* b2 = (const float*)d_in[4];
  float* out = (float*)d_out;

  char* ws = (char*)d_ws;
  float* casC = (float*)ws;                                    // 512*512 f32
  float* casW = (float*)(ws + (size_t)CC * CC * 4);            // 360*360 f32
  float* big  = (float*)(ws + (size_t)CC * CC * 4 + 524288);   // 64800*512 f32

  const long long PLANE = (long long)WW * CC;
  const long long TOTAL = (long long)MTOT * CC;

  build_cas_kernel<<<(CC * CC + 255) / 256, 256, 0, stream>>>(casC, CC);
  build_cas_kernel<<<(WW * WW + 255) / 256, 256, 0, stream>>>(casW, WW);

  dim3 blk(256);
  dim3 gBigC((MTOT + BM - 1) / BM, CC / BN, 1);        // 507 x 8
  dim3 gRowW((WW + BM - 1) / BM, CC / BN, HH);         // 3 x 8 x 180

  // 1) y1 = x @ casC            -> big          (B = casC symmetric)
  gemm_split_bf16<<<gBigC, blk, 0, stream>>>(
      x, 0, CC, casC, 0, CC, big, 0, CC, nullptr, 0, MTOT, CC, CC, 1.0f, 1);
  // 2) xh[h] = casW @ y1[h]     -> d_out
  gemm_split_bf16<<<gRowW, blk, 0, stream>>>(
      casW, 0, WW, big, PLANE, CC, out, PLANE, CC, nullptr, 0, WW, CC, WW, 1.0f, 0);
  // 3) fused block MLP + softshrink:  d_out(xh) -> big(y_mid)
  dim3 gMlp(MTOT / MT, NB, 1);                          // 2025 x 8
  afno_mlp_kernel<<<gMlp, blk, 0, stream>>>(out, w1, b1, w2, b2, big);
  // 4) z = y_mid @ casC         -> d_out        (B = casC symmetric)
  gemm_split_bf16<<<gBigC, blk, 0, stream>>>(
      big, 0, CC, casC, 0, CC, out, 0, CC, nullptr, 0, MTOT, CC, CC, 1.0f, 1);
  // 5) y[h] = casW @ z[h] / (W*C) + x[h]  -> big
  gemm_split_bf16<<<gRowW, blk, 0, stream>>>(
      casW, 0, WW, out, PLANE, CC, big, PLANE, CC, x, PLANE,
      WW, CC, WW, 1.0f / ((float)WW * (float)CC), 0);
  // 6) big -> d_out
  hipMemcpyAsync(out, big, (size_t)TOTAL * 4, hipMemcpyDeviceToDevice, stream);
}
